// VnDgcnnSE3_62457414419064
// MI455X (gfx1250) — matrix operand order repaired and checked by reading, not compiled, not run on hardware
//
#include <hip/hip_runtime.h>
#include <hip/hip_bf16.h>
#include <cstdint>

#define BATCH 8
#define NPTS 1024
#define KNN 20
#define EPSF 1e-6f
#define BN_EPSF 1e-5f
#define SLOPEF 0.2f

typedef _Float16 half_t;
typedef __attribute__((ext_vector_type(16))) _Float16 v16h;
typedef __attribute__((ext_vector_type(8)))  float    v8f;
typedef __attribute__((ext_vector_type(4)))  unsigned int v4u;
typedef __attribute__((ext_vector_type(8)))  int v8i_t;
typedef __attribute__((ext_vector_type(4)))  int v4i_t;

#if __has_builtin(__builtin_amdgcn_tensor_load_to_lds)
#define HAVE_TDM 1
#endif

// ---------------------------------------------------------------------------
// Tensor Data Mover: stage a padded f16 weight matrix [rows][cols] into LDS.
// D# built per CDNA5 ISA (group0/group1 bitfields).
// LDS byte address = low 32 bits of the flat shared pointer (ISA 10.2: LDS
// aperture address truncates to addr[31:0]).
// ---------------------------------------------------------------------------
__device__ __forceinline__ void tdm_stage_weights(
    const half_t* __restrict__ gsrc, half_t* lds_dst, int rows, int cols) {
#if defined(HAVE_TDM)
  unsigned lds_addr = (unsigned)(uintptr_t)lds_dst;
  unsigned long long ga = (unsigned long long)(uintptr_t)gsrc;
  v4u g0;
  g0[0] = 1u;                                  // count=1, user descriptor
  g0[1] = lds_addr;                            // lds_addr[31:0]
  g0[2] = (unsigned)ga;                        // global_addr[31:0]
  g0[3] = (unsigned)((ga >> 32) & 0x01FFFFFFu) // global_addr[56:32]
          | (2u << 30);                        // type=2 ("image")
  v8i_t g1;
  g1[0] = 0x10000;                   // workgroup_mask=0, data_size=1 (2 bytes)
  g1[1] = (cols & 0xFFFF) << 16;     // tensor_dim0[15:0] @ bits 63:48
  g1[2] = (rows & 0xFFFF) << 16;     // tensor_dim1[15:0] @ bits 95:80
  g1[3] = (cols & 0xFFFF) << 16;     // tile_dim0 @ bits 127:112
  g1[4] = rows & 0xFFFF;             // tile_dim1 @ bits 143:128, tile_dim2=0
  g1[5] = cols;                      // tensor_dim0_stride[31:0]
  g1[6] = 0;
  g1[7] = 0;
  v4i_t gz = {0, 0, 0, 0};
#if defined(__clang_major__) && __clang_major__ >= 23
  v8i_t gz8 = {0, 0, 0, 0, 0, 0, 0, 0};
  __builtin_amdgcn_tensor_load_to_lds(g0, g1, gz, gz, gz8, 0);
#else
  __builtin_amdgcn_tensor_load_to_lds(g0, g1, gz, gz, 0);
#endif
#else
  int lane = threadIdx.x & 31;
  for (int i = lane; i < rows * cols; i += 32) lds_dst[i] = gsrc[i];
#endif
}

__device__ __forceinline__ void tdm_wait() {
#if defined(HAVE_TDM) && __has_builtin(__builtin_amdgcn_s_wait_tensorcnt)
  __builtin_amdgcn_s_wait_tensorcnt(0);
#endif
}

// ---------------------------------------------------------------------------
// WMMA helpers (layouts per CDNA5 ISA 7.12.2, wave32)
// ---------------------------------------------------------------------------
__device__ __forceinline__ v8f wmma_acc(v16h a, v16h b, v8f c) {
  return __builtin_amdgcn_wmma_f32_16x16x32_f16(false, a, false, b, (short)0, c,
                                                false, false);
}

// A-matrix 16x32 f16 fragment from padded row-major W[Mpad][KPAD].
// lane<16: h0..15 -> K 0..15 (plus kt*32)
// lane>=16: h0..15 -> K 16..31
__device__ __forceinline__ v16h load_afrag(const half_t* __restrict__ W,
                                           int Kpad, int mt, int kt, int lane) {
  int row = mt * 16 + (lane & 15);
  int hi  = (lane >> 4) & 1;
  const half_t* p = W + (size_t)row * Kpad + kt * 32 + hi * 16;
  v16h a;
#pragma unroll
  for (int h = 0; h < 8; ++h)  a[h] = p[h];      // 16B at base
#pragma unroll
  for (int h = 8; h < 16; ++h) a[h] = p[h];      // 16B at +16B
  return a;
}

// One 16-channel slice of the DGCNN edge feature for this lane's column.
// KB and C are compile-time: the channel-region selection folds away.
template <int C, int KB>
__device__ __forceinline__ v16h gather16(const float* __restrict__ xb, int n,
                                         int j, int v) {
  v16h r;
#pragma unroll
  for (int h = 0; h < 16; ++h) {
    const int ci = KB + h;  // compile-time after unroll
    float val = 0.f;
    if (ci < C) {
      const float* row = xb + (size_t)(ci * 3 + v) * NPTS;
      val = row[j] - row[n];
    } else if (ci < 2 * C) {
      const float* row = xb + (size_t)((ci - C) * 3 + v) * NPTS;
      val = row[n];
    }
    r[h] = (half_t)val;
  }
  return r;
}

// Build all B fragments (3 vector components x KT k-tiles). One divergent
// branch on the lane half selects the K sub-range (contiguous k = 16*hi + h, as in load_afrag).
template <int C, int KT>
__device__ __forceinline__ void build_feat_all(v16h (&bf)[3][KT],
                                               const float* __restrict__ xb,
                                               int n, int j, int hi) {
  if (hi) {
#pragma unroll
    for (int v = 0; v < 3; ++v) {
      if constexpr (KT > 0) bf[v][0] = gather16<C, 16>(xb, n, j, v);
      if constexpr (KT > 1) bf[v][1] = gather16<C, 48>(xb, n, j, v);
      if constexpr (KT > 2) bf[v][2] = gather16<C, 80>(xb, n, j, v);
    }
  } else {
#pragma unroll
    for (int v = 0; v < 3; ++v) {
      if constexpr (KT > 0) bf[v][0] = gather16<C, 0>(xb, n, j, v);
      if constexpr (KT > 1) bf[v][1] = gather16<C, 32>(xb, n, j, v);
      if constexpr (KT > 2) bf[v][2] = gather16<C, 64>(xb, n, j, v);
    }
  }
}

// ---------------------------------------------------------------------------
// kNN: one thread per point, coords staged in LDS, register top-20
// ---------------------------------------------------------------------------
__global__ __launch_bounds__(256) void knn_kernel(const float* __restrict__ x,
                                                  int* __restrict__ idxout) {
  __shared__ float sx[3 * NPTS]; // 12 KB
  int tid = threadIdx.x;
  int b = blockIdx.x / (NPTS / 256);
  int chunk = blockIdx.x % (NPTS / 256);
  for (int i = tid; i < 3 * NPTS; i += 256)
    sx[i] = x[(size_t)b * 3 * NPTS + i];
  __syncthreads();

  int n = chunk * 256 + tid;
  float px = sx[0 * NPTS + n], py = sx[1 * NPTS + n], pz = sx[2 * NPTS + n];
  float bd[KNN];
  int bi[KNN];
#pragma unroll
  for (int t = 0; t < KNN; ++t) { bd[t] = 3.4e38f; bi[t] = 0; }

  for (int m = 0; m < NPTS; ++m) {
    float dx = sx[0 * NPTS + m] - px;
    float dy = sx[1 * NPTS + m] - py;
    float dz = sx[2 * NPTS + m] - pz;
    float d2 = dx * dx + dy * dy + dz * dz;
    float cd = d2; int cm = m;
#pragma unroll
    for (int t = 0; t < KNN; ++t) {
      if (cd < bd[t]) {
        float td = bd[t]; bd[t] = cd; cd = td;
        int ti = bi[t]; bi[t] = cm; cm = ti;
      }
    }
  }
#pragma unroll
  for (int t = 0; t < KNN; ++t)
    idxout[((size_t)b * NPTS + n) * KNN + t] = bi[t];
}

// ---------------------------------------------------------------------------
// BN statistics pass: Wf staged to LDS by the Tensor Data Mover, then
// p = Wf * feat via WMMA, accumulate sum/sumsq of ||p||.
// ---------------------------------------------------------------------------
template <int C, int Co, int Mt, int Kt, int KPAD>
__global__ __launch_bounds__(256) void stats_kernel(
    const float* __restrict__ xprev, const int* __restrict__ idx,
    const half_t* __restrict__ Wf16, float* __restrict__ gsum,
    float* __restrict__ gsq) {
  __shared__ half_t s_w[Mt * 16 * KPAD];
  __shared__ float s_sum[Co], s_sq[Co];
  int tid = threadIdx.x, lane = tid & 31, wave = tid >> 5;
  if (tid < Co) { s_sum[tid] = 0.f; s_sq[tid] = 0.f; }
  if (wave == 0) {
    tdm_stage_weights(Wf16, s_w, Mt * 16, KPAD); // async DMA global->LDS
    tdm_wait();                                  // s_wait_tensorcnt 0
  }
  __syncthreads();

  const int ctPerB = NPTS * KNN / 16; // 1280
  int gct = blockIdx.x * 8 + wave;
  int b = gct / ctPerB, ct = gct % ctPerB;
  const float* xb = xprev + (size_t)b * C * 3 * NPTS;
  __builtin_prefetch(xb, 0, 0);

  int col = ct * 16 + (lane & 15);
  int n = col / KNN, kk = col % KNN;
  int hi = (lane >> 4) & 1;
  int j = idx[((size_t)b * NPTS + n) * KNN + kk];

  v16h bf[3][Kt];
  build_feat_all<C, Kt>(bf, xb, n, j, hi);

#pragma unroll
  for (int mt = 0; mt < Mt; ++mt) {
    v8f acc[3] = {};
#pragma unroll
    for (int kt = 0; kt < Kt; ++kt) {
      v16h a = load_afrag(s_w, KPAD, mt, kt, lane); // ds reads
#pragma unroll
      for (int v = 0; v < 3; ++v) acc[v] = wmma_acc(a, bf[v][kt], acc[v]);
    }
#pragma unroll
    for (int r = 0; r < 8; ++r) {
      int co = mt * 16 + hi * 8 + r;
      if (co < Co) {
        float p0 = acc[0][r], p1 = acc[1][r], p2 = acc[2][r];
        float nrm = sqrtf(p0 * p0 + p1 * p1 + p2 * p2) + EPSF;
        atomicAdd(&s_sum[co], nrm);
        atomicAdd(&s_sq[co], nrm * nrm);
      }
    }
  }
  __syncthreads();
  if (tid < Co) {
    atomicAdd(&gsum[tid], s_sum[tid]);
    atomicAdd(&gsq[tid], s_sq[tid]);
  }
}

__global__ void finalize_kernel(const float* __restrict__ sum,
                                const float* __restrict__ sq,
                                float* __restrict__ mean,
                                float* __restrict__ istd, int Co) {
  int t = threadIdx.x;
  if (t < Co) {
    const float cnt = (float)(BATCH * NPTS * KNN);
    float mu = sum[t] / cnt;
    float var = sq[t] / cnt - mu * mu;
    mean[t] = mu;
    istd[t] = rsqrtf(var + BN_EPSF);
  }
}

// ---------------------------------------------------------------------------
// Fused apply: recompute p=Wf*feat, d=Wd*feat, BN + VN-LeakyReLU -> LDS f16
// (layout [col][v][ci] so Wp B-fragments are contiguous ds reads), then
// d2 = Wp * act via WMMA, dot, argmax over k, write pooled f32 output.
// ---------------------------------------------------------------------------
template <int C, int Co, int Mt, int Kt, int KPAD, int Kt2, int KPAD2, int TN>
__global__ __launch_bounds__(256) void apply_kernel(
    const float* __restrict__ xprev, const int* __restrict__ idx,
    const half_t* __restrict__ Wf16, const half_t* __restrict__ Wd16,
    const half_t* __restrict__ Wp16, const float* __restrict__ mean,
    const float* __restrict__ istd, const float* __restrict__ gamma,
    const float* __restrict__ beta, float* __restrict__ xout) {
  constexpr int COLSL = TN * KNN;
  constexpr int CTSL = COLSL / 16;
  __shared__ half_t x_lds[COLSL * 3 * KPAD2]; // [col][v][ci]
  __shared__ half_t dot_lds[COLSL * Co];      // [col][co]

  int tid = threadIdx.x, lane = tid & 31, wave = tid >> 5;
  int b = blockIdx.x / (NPTS / TN);
  int n0 = (blockIdx.x % (NPTS / TN)) * TN;
  const float* xb = xprev + (size_t)b * C * 3 * NPTS;
  __builtin_prefetch(xb, 0, 0);

  for (int i = tid; i < COLSL * 3 * KPAD2; i += 256) x_lds[i] = (half_t)0.f;
  __syncthreads();

  int hi = (lane >> 4) & 1;
  for (int ct = wave; ct < CTSL; ct += 8) {
    int col = ct * 16 + (lane & 15);
    int nl = col / KNN, kk = col % KNN;
    int n = n0 + nl;
    int j = idx[((size_t)b * NPTS + n) * KNN + kk];

    v16h bf[3][Kt];
    build_feat_all<C, Kt>(bf, xb, n, j, hi);

#pragma unroll
    for (int mt = 0; mt < Mt; ++mt) {
      v8f ap[3] = {}, ad[3] = {};
#pragma unroll
      for (int kt = 0; kt < Kt; ++kt) {
        v16h af  = load_afrag(Wf16, KPAD, mt, kt, lane);
        v16h adg = load_afrag(Wd16, KPAD, mt, kt, lane);
#pragma unroll
        for (int v = 0; v < 3; ++v) {
          ap[v] = wmma_acc(af, bf[v][kt], ap[v]);
          ad[v] = wmma_acc(adg, bf[v][kt], ad[v]);
        }
      }
      // BN on ||p|| + vector-neuron LeakyReLU; store act to [col][v][co]
#pragma unroll
      for (int r = 0; r < 8; ++r) {
        int co = mt * 16 + hi * 8 + r;
        if (co < Co) {
          float p0 = ap[0][r], p1 = ap[1][r], p2 = ap[2][r];
          float d0 = ad[0][r], d1 = ad[1][r], d2 = ad[2][r];
          float nrm = sqrtf(p0 * p0 + p1 * p1 + p2 * p2) + EPSF;
          float nbn = gamma[co] * (nrm - mean[co]) * istd[co] + beta[co];
          float s = nbn / nrm;
          p0 *= s; p1 *= s; p2 *= s;
          float dot = p0 * d0 + p1 * d1 + p2 * d2;
          float dsq = d0 * d0 + d1 * d1 + d2 * d2;
          float t = dot / (dsq + EPSF);
          float q0 = (dot >= 0.f) ? p0 : (p0 - t * d0);
          float q1 = (dot >= 0.f) ? p1 : (p1 - t * d1);
          float q2 = (dot >= 0.f) ? p2 : (p2 - t * d2);
          x_lds[(col * 3 + 0) * KPAD2 + co] =
              (half_t)(SLOPEF * p0 + (1.f - SLOPEF) * q0);
          x_lds[(col * 3 + 1) * KPAD2 + co] =
              (half_t)(SLOPEF * p1 + (1.f - SLOPEF) * q1);
          x_lds[(col * 3 + 2) * KPAD2 + co] =
              (half_t)(SLOPEF * p2 + (1.f - SLOPEF) * q2);
        }
      }
    }

    // Wp B fragments: 16 consecutive halfs per lane (conflict-free ds reads)
    v16h bx[3][Kt2];
#pragma unroll
    for (int v = 0; v < 3; ++v)
#pragma unroll
      for (int kt = 0; kt < Kt2; ++kt) {
        int kb = kt * 32 + hi * 16;
        v16h t;
#pragma unroll
        for (int h = 0; h < 16; ++h)
          t[h] = x_lds[(col * 3 + v) * KPAD2 + kb + h];
        bx[v][kt] = t;
      }

#pragma unroll
    for (int mt = 0; mt < Mt; ++mt) {
      v8f a2[3] = {};
#pragma unroll
      for (int kt = 0; kt < Kt2; ++kt) {
        v16h a = load_afrag(Wp16, KPAD2, mt, kt, lane);
#pragma unroll
        for (int v = 0; v < 3; ++v) a2[v] = wmma_acc(a, bx[v][kt], a2[v]);
      }
#pragma unroll
      for (int r = 0; r < 8; ++r) {
        int co = mt * 16 + hi * 8 + r;
        if (co < Co) {
          float xv0 = (float)x_lds[(col * 3 + 0) * KPAD2 + co];
          float xv1 = (float)x_lds[(col * 3 + 1) * KPAD2 + co];
          float xv2 = (float)x_lds[(col * 3 + 2) * KPAD2 + co];
          float dd = xv0 * a2[0][r] + xv1 * a2[1][r] + xv2 * a2[2][r];
          dot_lds[col * Co + co] = (half_t)dd;
        }
      }
    }
  }
  __syncthreads();

  // pool over k (argmax of dot, first max wins), write f32 output
  for (int e = tid; e < Co * TN; e += 256) {
    int co = e / TN, nl = e % TN;
    int bestk = 0;
    float bestv = -3.0e38f;
#pragma unroll
    for (int k = 0; k < KNN; ++k) {
      float dv = (float)dot_lds[(nl * KNN + k) * Co + co];
      if (dv > bestv) { bestv = dv; bestk = k; }
    }
    int n = n0 + nl;
#pragma unroll
    for (int v = 0; v < 3; ++v)
      xout[(((size_t)b * Co + co) * 3 + v) * NPTS + n] =
          (float)x_lds[((nl * KNN + bestk) * 3 + v) * KPAD2 + co];
  }
}

// ---------------------------------------------------------------------------
// utility kernels
// ---------------------------------------------------------------------------
__global__ void wconv_kernel(const float* __restrict__ src,
                             half_t* __restrict__ dst, int Co, int Ci,
                             int Kpad, int total) {
  int t = blockIdx.x * 256 + threadIdx.x;
  if (t >= total) return;
  int m = t / Kpad, k = t % Kpad;
  float v = (m < Co && k < Ci) ? src[(size_t)m * Ci + k] : 0.f;
  dst[t] = (half_t)v;
}

__global__ void zero_kernel(float* p, int n) {
  int t = blockIdx.x * 256 + threadIdx.x;
  if (t < n) p[t] = 0.f;
}

__global__ void concat_kernel(const float* __restrict__ src,
                              float* __restrict__ dst, int Co, int coOff,
                              int CoTot, int total) {
  int t = blockIdx.x * 256 + threadIdx.x;
  if (t >= total) return;
  int n = t % NPTS;
  int v = (t / NPTS) % 3;
  int co = (t / (3 * NPTS)) % Co;
  int b = t / (Co * 3 * NPTS);
  dst[(((size_t)b * CoTot + coOff + co) * 3 + v) * NPTS + n] = src[t];
}

// ---------------------------------------------------------------------------
template <int C, int Co, int Mt, int Kt, int KPAD, int Kt2, int KPAD2, int TN>
static void run_layer(const float* xprev, const int* idxp, const half_t* wf,
                      const half_t* wd, const half_t* wp, float* sum, float* sq,
                      float* mu, float* is, const float* g, const float* be,
                      float* xout, hipStream_t stream) {
  stats_kernel<C, Co, Mt, Kt, KPAD>
      <<<BATCH * (NPTS * KNN / 16) / 8, 256, 0, stream>>>(xprev, idxp, wf, sum,
                                                          sq);
  finalize_kernel<<<1, 96, 0, stream>>>(sum, sq, mu, is, Co);
  apply_kernel<C, Co, Mt, Kt, KPAD, Kt2, KPAD2, TN>
      <<<BATCH * (NPTS / TN), 256, 0, stream>>>(xprev, idxp, wf, wd, wp, mu, is,
                                                g, be, xout);
}

extern "C" void kernel_launch(void* const* d_in, const int* in_sizes, int n_in,
                              void* d_out, int out_size, void* d_ws,
                              size_t ws_size, hipStream_t stream) {
  (void)in_sizes; (void)n_in; (void)out_size; (void)ws_size;
  static const int Cl[4]     = {1, 21, 21, 42};
  static const int Col[4]    = {21, 21, 42, 85};
  static const int Mtl[4]    = {2, 2, 3, 6};
  static const int Kpadl[4]  = {32, 64, 64, 96};
  static const int Kpad2l[4] = {32, 32, 64, 96};

  const float* x = (const float*)d_in[0];
  const float* Wf[4]; const float* Wd[4]; const float* g[4];
  const float* be[4]; const float* Wp[4];
  for (int l = 0; l < 4; ++l) {
    Wf[l] = (const float*)d_in[1 + 5 * l + 0];
    Wd[l] = (const float*)d_in[1 + 5 * l + 1];
    g[l]  = (const float*)d_in[1 + 5 * l + 2];
    be[l] = (const float*)d_in[1 + 5 * l + 3];
    Wp[l] = (const float*)d_in[1 + 5 * l + 4];
  }

  char* ws = (char*)d_ws;
  size_t off = 0;
  auto carve = [&](size_t bytes) {
    size_t o = off;
    off += (bytes + 255) & ~(size_t)255;
    return o;
  };
  size_t idx_off = carve((size_t)BATCH * NPTS * KNN * sizeof(int));
  size_t wf_off[4], wd_off[4], wp_off[4], xbuf_off[4];
  for (int l = 0; l < 4; ++l) {
    wf_off[l] = carve((size_t)Mtl[l] * 16 * Kpadl[l] * sizeof(half_t));
    wd_off[l] = carve((size_t)Mtl[l] * 16 * Kpadl[l] * sizeof(half_t));
    wp_off[l] = carve((size_t)Mtl[l] * 16 * Kpad2l[l] * sizeof(half_t));
  }
  size_t stats_off = carve((size_t)4 * 384 * sizeof(float));
  for (int l = 0; l < 4; ++l)
    xbuf_off[l] = carve((size_t)BATCH * Col[l] * 3 * NPTS * sizeof(float));

  int* idxp = (int*)(ws + idx_off);
  float* stats = (float*)(ws + stats_off);

  // zero BN accumulators (ws is poisoned; must re-zero every call)
  zero_kernel<<<(4 * 384 + 255) / 256, 256, 0, stream>>>(stats, 4 * 384);

  // convert + pad weights to f16
  for (int l = 0; l < 4; ++l) {
    int Ci = 2 * Cl[l];
    int tf = Mtl[l] * 16 * Kpadl[l];
    wconv_kernel<<<(tf + 255) / 256, 256, 0, stream>>>(
        Wf[l], (half_t*)(ws + wf_off[l]), Col[l], Ci, Kpadl[l], tf);
    wconv_kernel<<<(tf + 255) / 256, 256, 0, stream>>>(
        Wd[l], (half_t*)(ws + wd_off[l]), Col[l], Ci, Kpadl[l], tf);
    int tp = Mtl[l] * 16 * Kpad2l[l];
    wconv_kernel<<<(tp + 255) / 256, 256, 0, stream>>>(
        Wp[l], (half_t*)(ws + wp_off[l]), Col[l], Col[l], Kpad2l[l], tp);
  }

  knn_kernel<<<BATCH * (NPTS / 256), 256, 0, stream>>>(x, idxp);

  float* st[4][4];
  for (int l = 0; l < 4; ++l)
    for (int q = 0; q < 4; ++q) st[l][q] = stats + l * 384 + q * 96;

  // layer 1: C=1  Co=21 Mt=2 Kt=1 KPAD=32 Kt2=1 KPAD2=32 TN=8
  run_layer<1, 21, 2, 1, 32, 1, 32, 8>(
      x, idxp, (const half_t*)(ws + wf_off[0]), (const half_t*)(ws + wd_off[0]),
      (const half_t*)(ws + wp_off[0]), st[0][0], st[0][1], st[0][2], st[0][3],
      g[0], be[0], (float*)(ws + xbuf_off[0]), stream);
  // layer 2: C=21 Co=21 Mt=2 Kt=2 KPAD=64 Kt2=1 KPAD2=32 TN=8
  run_layer<21, 21, 2, 2, 64, 1, 32, 8>(
      (const float*)(ws + xbuf_off[0]), idxp, (const half_t*)(ws + wf_off[1]),
      (const half_t*)(ws + wd_off[1]), (const half_t*)(ws + wp_off[1]),
      st[1][0], st[1][1], st[1][2], st[1][3], g[1], be[1],
      (float*)(ws + xbuf_off[1]), stream);
  // layer 3: C=21 Co=42 Mt=3 Kt=2 KPAD=64 Kt2=2 KPAD2=64 TN=4
  run_layer<21, 42, 3, 2, 64, 2, 64, 4>(
      (const float*)(ws + xbuf_off[1]), idxp, (const half_t*)(ws + wf_off[2]),
      (const half_t*)(ws + wd_off[2]), (const half_t*)(ws + wp_off[2]),
      st[2][0], st[2][1], st[2][2], st[2][3], g[2], be[2],
      (float*)(ws + xbuf_off[2]), stream);
  // layer 4: C=42 Co=85 Mt=6 Kt=3 KPAD=96 Kt2=3 KPAD2=96 TN=4
  run_layer<42, 85, 6, 3, 96, 3, 96, 4>(
      (const float*)(ws + xbuf_off[2]), idxp, (const half_t*)(ws + wf_off[3]),
      (const half_t*)(ws + wd_off[3]), (const half_t*)(ws + wp_off[3]),
      st[3][0], st[3][1], st[3][2], st[3][3], g[3], be[3],
      (float*)(ws + xbuf_off[3]), stream);

  // outputs: (x4, pyramid concat, coords)
  float* out = (float*)d_out;
  const size_t X4_ELEMS = (size_t)BATCH * 85 * 3 * NPTS;   // 2088960
  const size_t PYR_ELEMS = (size_t)BATCH * 169 * 3 * NPTS; // 4153344
  hipMemcpyAsync(out, ws + xbuf_off[3], X4_ELEMS * sizeof(float),
                 hipMemcpyDeviceToDevice, stream);
  int coOff[4] = {0, 21, 42, 84};
  static const int ColH[4] = {21, 21, 42, 85};
  for (int l = 0; l < 4; ++l) {
    int total = BATCH * ColH[l] * 3 * NPTS;
    concat_kernel<<<(total + 255) / 256, 256, 0, stream>>>(
        (const float*)(ws + xbuf_off[l]), out + X4_ELEMS, ColH[l], coOff[l],
        169, total);
  }
  hipMemcpyAsync(out + X4_ELEMS + PYR_ELEMS, x,
                 (size_t)BATCH * 3 * NPTS * sizeof(float),
                 hipMemcpyDeviceToDevice, stream);
}
